// VanillaLSTMModel_68058051772555
// MI455X (gfx1250) — compile-verified, matrix operand order checked by reading, not run on hardware
//
#include <hip/hip_runtime.h>
#include <hip/hip_bf16.h>

// ---------------- problem constants ----------------
static constexpr int H     = 512;    // hidden
static constexpr int FOURH = 2048;   // 4*H
static constexpr int II    = 256;    // intermediate
static constexpr int DIN   = 2;
static constexpr int DOUT  = 5;
static constexpr int NN    = 1024;   // batch
static constexpr int TT    = 128;    // steps
static constexpr int MROWS = NN * TT; // 131072 head rows

// ---------------- types ----------------
typedef __attribute__((ext_vector_type(16))) __bf16 v16bf;
typedef __attribute__((ext_vector_type(8)))  __bf16 v8bf;
typedef __attribute__((ext_vector_type(8)))  float  v8f;

union BF16x16 { v16bf v; v8bf h[2]; };

// ---------------- workspace layout (bytes) ----------------
static constexpr size_t OFF_BAR  = 0;                                   // barrier counter
static constexpr size_t OFF_HBF  = 256;                                 // h state, bf16, double buffered: 2*N*H*2 = 2 MB
static constexpr size_t OFF_C    = OFF_HBF + (size_t)2 * NN * H * 2;    // c state fp32: 2 MB
static constexpr size_t OFF_WB   = OFF_C   + (size_t)NN * H * 4;        // w_hh bf16 swizzled: 2 MB
static constexpr size_t OFF_W1B  = OFF_WB  + (size_t)FOURH * H * 2;     // w1 bf16 swizzled: 256 KB
static constexpr size_t OFF_HS   = OFF_W1B + (size_t)II * H * 2;        // hs bf16 [T*N, H]: 128 MB
static constexpr size_t OFF_ACT  = OFF_HS  + (size_t)MROWS * H * 2;     // relu acts bf16 [T*N, I]: 64 MB

// fragment-layout helper: element (n_or_m = nl, k = kl) of a 16x32 bf16 tile
// lives at lane = nl + 16*((kl>>3)&1), slot v = 4*(kl>=16) + ((kl&7)>>1),
// i.e. bf16 index within the 512-element fragment = lane*16 + v*2 + (kl&1).
__device__ __forceinline__ int frag_elem_idx(int nl, int kl) {
    int lane = nl + ((kl >> 3) & 1) * 16;
    int v    = ((kl >> 4) & 1) * 4 + ((kl & 7) >> 1);
    return lane * 16 + v * 2 + (kl & 1);
}

__device__ __forceinline__ float sigm(float x) { return 1.0f / (1.0f + __expf(-x)); }

// ---------------- prep: zero state, reset barrier, swizzle weights to bf16 fragments ----------------
__global__ __launch_bounds__(256) void prep_kernel(const float* __restrict__ w_hh,
                                                   const float* __restrict__ w1,
                                                   __bf16* __restrict__ hbf,
                                                   float*  __restrict__ c,
                                                   __bf16* __restrict__ Wb,
                                                   __bf16* __restrict__ W1b,
                                                   unsigned* __restrict__ bar) {
    int idx = blockIdx.x * 256 + threadIdx.x;
    if (idx == 0) *bar = 0u;

    if (idx < 2 * NN * H) {                       // zero both h buffers
        hbf[idx] = (__bf16)0.0f;
        if (idx < NN * H) c[idx] = 0.0f;          // zero c
        return;
    }
    int e = idx - 2 * NN * H;
    if (e < FOURH * H) {                          // swizzle w_hh -> Wb
        int grow = e >> 9;        // row of w_hh (gate-major, 0..2047)
        int k    = e & (H - 1);
        int g    = grow >> 9;     // gate 0..3
        int gc   = grow & (H - 1);
        int ct   = gc >> 5;       // col tile 0..15
        int wn   = (gc >> 4) & 1; // col half
        int nl   = gc & 15;
        int kc   = k >> 5;
        int kl   = k & 31;
        int frag = ((ct * 16 + kc) * 4 + g) * 2 + wn;
        Wb[(size_t)frag * 512 + frag_elem_idx(nl, kl)] = (__bf16)w_hh[e];
        return;
    }
    e -= FOURH * H;
    if (e < II * H) {                             // swizzle w1 -> W1b
        int col = e >> 9;
        int k   = e & (H - 1);
        int nf  = col >> 4;
        int nl  = col & 15;
        int kc  = k >> 5;
        int kl  = k & 31;
        W1b[(size_t)(nf * 16 + kc) * 512 + frag_elem_idx(nl, kl)] = (__bf16)w1[e];
    }
}

// ---------------- persistent LSTM: all T steps, grid-wide barrier between steps ----------------
// grid = (8 row-tiles, 16 col-tiles), block = 256 (8 waves: 4 M x 2 N-half)
__global__ __launch_bounds__(256) void lstm_persist_kernel(const float* __restrict__ x,
                                                           const float* __restrict__ w_ih,
                                                           const float* __restrict__ b_ih,
                                                           const float* __restrict__ b_hh,
                                                           const __bf16* __restrict__ Wb,
                                                           __bf16* __restrict__ hbf,
                                                           float*  __restrict__ c,
                                                           __bf16* __restrict__ hsbf,
                                                           unsigned* __restrict__ bar) {
    const int tid   = threadIdx.x;
    const int lane  = tid & 31;
    const int wave  = tid >> 5;
    const int wm    = wave >> 1;   // 0..3 : which 32-row sub-tile
    const int wn    = wave & 1;    // 0..1 : which 16-col half
    const int rt    = blockIdx.x;  // 0..7
    const int ct    = blockIdx.y;  // 0..15
    const int laneN = lane & 15;
    const int laneH = lane >> 4;
    const int K0    = laneH * 8;   // A/B fragment k-offset for this lane
    const int rowbase = rt * 128 + wm * 32;
    const int hcol    = ct * 32 + wn * 16 + laneN;

    // per-lane gate constants (invariant over t)
    float bias[4], wa[4], wb_[4];
#pragma unroll
    for (int g = 0; g < 4; ++g) {
        int j   = g * H + hcol;
        bias[g] = b_ih[j] + b_hh[j];
        wa[g]   = w_ih[2 * j + 0];
        wb_[g]  = w_ih[2 * j + 1];
    }

    const unsigned nwg = gridDim.x * gridDim.y;   // 128

    for (int t = 0; t < TT; ++t) {
        const __bf16* hr = hbf + (size_t)(t & 1) * NN * H;        // read buffer
        __bf16*       hw = hbf + (size_t)((t + 1) & 1) * NN * H;  // write buffer

        v8f acc[2][4];
#pragma unroll
        for (int mf = 0; mf < 2; ++mf)
#pragma unroll
            for (int g = 0; g < 4; ++g) acc[mf][g] = (v8f)0.0f;

#pragma unroll 4
        for (int kc = 0; kc < 16; ++kc) {
            BF16x16 a[2], b[4];
#pragma unroll
            for (int mf = 0; mf < 2; ++mf) {
                const __bf16* p = hr + (size_t)(rowbase + mf * 16 + laneN) * H + kc * 32 + K0;
                a[mf].h[0] = *(const v8bf*)(p);
                a[mf].h[1] = *(const v8bf*)(p + 16);
            }
#pragma unroll
            for (int g = 0; g < 4; ++g) {
                const __bf16* p = Wb + (size_t)(((ct * 16 + kc) * 4 + g) * 2 + wn) * 512 + lane * 16;
                b[g].h[0] = *(const v8bf*)(p);
                b[g].h[1] = *(const v8bf*)(p + 8);
            }
#pragma unroll
            for (int mf = 0; mf < 2; ++mf)
#pragma unroll
                for (int g = 0; g < 4; ++g)
                    acc[mf][g] = __builtin_amdgcn_wmma_f32_16x16x32_bf16(
                        false, a[mf].v, false, b[g].v, (short)0, acc[mf][g], false, false);
        }

        // fused pointwise: all four gates for (row, hcol) are resident in this lane
#pragma unroll
        for (int mf = 0; mf < 2; ++mf) {
#pragma unroll
            for (int v = 0; v < 8; ++v) {
                int row = rowbase + mf * 16 + laneH * 8 + v;
                const float* xp = x + ((size_t)row * TT + t) * DIN;
                float x0 = xp[0], x1 = xp[1];
                float gi = acc[mf][0][v] + bias[0] + x0 * wa[0] + x1 * wb_[0];
                float gf = acc[mf][1][v] + bias[1] + x0 * wa[1] + x1 * wb_[1];
                float gg = acc[mf][2][v] + bias[2] + x0 * wa[2] + x1 * wb_[2];
                float go = acc[mf][3][v] + bias[3] + x0 * wa[3] + x1 * wb_[3];
                float iv = sigm(gi), fv = sigm(gf), gv = tanhf(gg), ov = sigm(go);
                size_t ci = (size_t)row * H + hcol;
                float cn = fv * c[ci] + iv * gv;
                c[ci] = cn;
                float hn = ov * tanhf(cn);
                __bf16 hb = (__bf16)hn;
                hw[ci] = hb;
                hsbf[(size_t)t * NN * H + ci] = hb;
            }
        }

        // device-wide step barrier (monotone counter; reset by prep each launch)
        __threadfence();
        __syncthreads();
        if (tid == 0) {
            __hip_atomic_fetch_add(bar, 1u, __ATOMIC_RELEASE, __HIP_MEMORY_SCOPE_AGENT);
            unsigned target = nwg * (unsigned)(t + 1);
            while (__hip_atomic_load(bar, __ATOMIC_ACQUIRE, __HIP_MEMORY_SCOPE_AGENT) < target)
                __builtin_amdgcn_s_sleep(1);
        }
        __syncthreads();
        __builtin_amdgcn_fence(__ATOMIC_ACQUIRE, "agent");
    }
}

// ---------------- head stage 1: act = relu(hs @ w1.T + b1), bf16 WMMA ----------------
// grid = (MROWS/128, 2), block = 256 (8 waves: 4 M x 2 N-half of the 128-col half)
__global__ __launch_bounds__(256) void head1_kernel(const __bf16* __restrict__ hsbf,
                                                    const __bf16* __restrict__ W1b,
                                                    const float* __restrict__ b1,
                                                    __bf16* __restrict__ act) {
    const int tid   = threadIdx.x;
    const int lane  = tid & 31;
    const int wave  = tid >> 5;
    const int wm    = wave >> 1;
    const int wn    = wave & 1;
    const int laneN = lane & 15;
    const int laneH = lane >> 4;
    const int K0    = laneH * 8;
    const int rowbase = blockIdx.x * 128 + wm * 32;
    const int colbase = blockIdx.y * 128 + wn * 64;

    v8f acc[2][4];
#pragma unroll
    for (int mf = 0; mf < 2; ++mf)
#pragma unroll
        for (int nb = 0; nb < 4; ++nb) acc[mf][nb] = (v8f)0.0f;

#pragma unroll 4
    for (int kc = 0; kc < 16; ++kc) {
        BF16x16 a[2], b[4];
#pragma unroll
        for (int mf = 0; mf < 2; ++mf) {
            const __bf16* p = hsbf + (size_t)(rowbase + mf * 16 + laneN) * H + kc * 32 + K0;
            a[mf].h[0] = *(const v8bf*)(p);
            a[mf].h[1] = *(const v8bf*)(p + 16);
        }
#pragma unroll
        for (int nb = 0; nb < 4; ++nb) {
            int nf = (colbase >> 4) + nb;
            const __bf16* p = W1b + (size_t)(nf * 16 + kc) * 512 + lane * 16;
            b[nb].h[0] = *(const v8bf*)(p);
            b[nb].h[1] = *(const v8bf*)(p + 8);
        }
#pragma unroll
        for (int mf = 0; mf < 2; ++mf)
#pragma unroll
            for (int nb = 0; nb < 4; ++nb)
                acc[mf][nb] = __builtin_amdgcn_wmma_f32_16x16x32_bf16(
                    false, a[mf].v, false, b[nb].v, (short)0, acc[mf][nb], false, false);
    }

#pragma unroll
    for (int nb = 0; nb < 4; ++nb) {
        int col   = colbase + nb * 16 + laneN;
        float bia = b1[col];
#pragma unroll
        for (int mf = 0; mf < 2; ++mf)
#pragma unroll
            for (int v = 0; v < 8; ++v) {
                int row = rowbase + mf * 16 + laneH * 8 + v;
                float val = acc[mf][nb][v] + bia;
                act[(size_t)row * II + col] = (__bf16)fmaxf(val, 0.0f);
            }
    }
}

// ---------------- head stage 2: out = act @ w2.T + b2 (memory-bound, w2 in LDS) ----------------
__global__ __launch_bounds__(256) void head2_kernel(const __bf16* __restrict__ act,
                                                    const float* __restrict__ w2,
                                                    const float* __restrict__ b2,
                                                    float* __restrict__ out) {
    __shared__ float w2s[DOUT * II];
    for (int i = threadIdx.x; i < DOUT * II; i += 256) w2s[i] = w2[i];
    __syncthreads();

    int row = blockIdx.x * 256 + threadIdx.x;
    if (row >= MROWS) return;

    float a0 = b2[0], a1 = b2[1], a2 = b2[2], a3 = b2[3], a4 = b2[4];
    const __bf16* ap = act + (size_t)row * II;
#pragma unroll 4
    for (int ii = 0; ii < II; ii += 8) {
        v8bf av = *(const v8bf*)(ap + ii);
#pragma unroll
        for (int e = 0; e < 8; ++e) {
            float a = (float)av[e];
            int i = ii + e;
            a0 += a * w2s[0 * II + i];
            a1 += a * w2s[1 * II + i];
            a2 += a * w2s[2 * II + i];
            a3 += a * w2s[3 * II + i];
            a4 += a * w2s[4 * II + i];
        }
    }
    float* op = out + (size_t)row * DOUT;
    op[0] = a0; op[1] = a1; op[2] = a2; op[3] = a3; op[4] = a4;
}

// ---------------- launch ----------------
extern "C" void kernel_launch(void* const* d_in, const int* in_sizes, int n_in,
                              void* d_out, int out_size, void* d_ws, size_t ws_size,
                              hipStream_t stream) {
    const float* x    = (const float*)d_in[0];
    const float* w_ih = (const float*)d_in[1];
    const float* w_hh = (const float*)d_in[2];
    const float* b_ih = (const float*)d_in[3];
    const float* b_hh = (const float*)d_in[4];
    const float* w1   = (const float*)d_in[5];
    const float* b1   = (const float*)d_in[6];
    const float* w2   = (const float*)d_in[7];
    const float* b2   = (const float*)d_in[8];
    float* out = (float*)d_out;

    char* base = (char*)d_ws;
    unsigned* bar  = (unsigned*)(base + OFF_BAR);
    __bf16*   hbf  = (__bf16*)(base + OFF_HBF);
    float*    c    = (float*)(base + OFF_C);
    __bf16*   Wb   = (__bf16*)(base + OFF_WB);
    __bf16*   W1b  = (__bf16*)(base + OFF_W1B);
    __bf16*   hsbf = (__bf16*)(base + OFF_HS);
    __bf16*   act  = (__bf16*)(base + OFF_ACT);

    // prep covers: 2*N*H h-zeros + N*H c-zeros (overlapped) + 4H*H + I*H swizzles
    int prep_items = 2 * NN * H + FOURH * H + II * H;
    prep_kernel<<<(prep_items + 255) / 256, 256, 0, stream>>>(w_hh, w1, hbf, c, Wb, W1b, bar);

    lstm_persist_kernel<<<dim3(8, 16), 256, 0, stream>>>(x, w_ih, b_ih, b_hh, Wb, hbf, c, hsbf, bar);

    head1_kernel<<<dim3(MROWS / 128, 2), 256, 0, stream>>>(hsbf, W1b, b1, act);

    head2_kernel<<<(MROWS + 255) / 256, 256, 0, stream>>>(act, w2, b2, out);

    (void)in_sizes; (void)n_in; (void)out_size; (void)ws_size;
}
